// lstm2_24318104830664
// MI455X (gfx1250) — matrix operand order repaired and checked by reading, not compiled, not run on hardware
//
#include <hip/hip_runtime.h>
#include <hip/hip_bf16.h>
#include <math.h>

#define B_ 64
#define T_ 1024
#define W_ 128
#define H_ 512
#define C_ 256
#define FORGET_BIAS 1.0f

typedef __attribute__((ext_vector_type(16))) __bf16   v16bf;
typedef __attribute__((ext_vector_type(8)))  float    v8f;
typedef __attribute__((ext_vector_type(4)))  unsigned v4u;

// ---------- helpers ----------

__device__ __forceinline__ unsigned short f2bfu(float f) {
  union { float f; unsigned u; } v; v.f = f;
  unsigned u = v.u;
  unsigned r = u + 0x7FFFu + ((u >> 16) & 1u);   // round-to-nearest-even
  return (unsigned short)(r >> 16);
}

__device__ __forceinline__ float sigf(float x) { return 1.0f / (1.0f + __expf(-x)); }

// A-matrix 16x32 bf16 lane layout (ISA 7.12.2):
// lanes 0-15: M=lane, VGPR0-3 hold K pairs 0..7, VGPR4-7 hold K 16..23
// lanes 16-31: same M, K offset +8
__device__ __forceinline__ int a_kpair(int lane, int r) {
  return ((r < 4) ? (2 * r) : (16 + 2 * (r - 4))) + ((lane >= 16) ? 8 : 0);
}

// Fragments live in memory as 512 contiguous ushorts, lane-major (lane*16
// elements = that lane's v16bf register image) -> two b128 loads.
// `base` is already lane-adjusted; `fragoff` is a compile-time constant after
// unrolling, so it folds into the global-load immediate offset.
__device__ __forceinline__ v16bf load_frag_off(const unsigned short* __restrict__ base,
                                               int fragoff) {
  union { v4u q[2]; v16bf v; } u;
  const v4u* q = reinterpret_cast<const v4u*>(base + ((size_t)fragoff << 9));
  u.q[0] = q[0];
  u.q[1] = q[1];
  return u.v;
}

// ---------- weight pre-swizzle: fp32 [K,N] -> bf16 WMMA B-fragment layout ----------
// B-matrix 32x16 bf16 layout: lane = N%16 (+16 for upper half), lanes0-15
// K=0..7,16..23 (2 per VGPR), lanes16-31 K=8..15,24..31. frag = ntile*ksteps + kstep.
__global__ void pack_b_kernel(const float* __restrict__ Wk, unsigned short* __restrict__ out,
                              int K, int N, int ksteps) {
  int id = blockIdx.x * blockDim.x + threadIdx.x;
  int total = (N >> 4) * ksteps * 512;
  if (id >= total) return;
  int within = id & 511;
  int frag = id >> 9;
  int ks = frag % ksteps;
  int nt = frag / ksteps;
  int lane = within >> 4;
  int e = within & 15;
  int r = e >> 1, sub = e & 1;
  int k = ks * 32 + ((lane >= 16) ? 8 : 0) + ((e < 8) ? e : (8 + e)) + 0 * (r + sub);
  int n = nt * 16 + (lane & 15);
  out[id] = f2bfu(Wk[(size_t)k * N + n]);
}

// ---------- x pre-swizzle: fp32 [B,T,W] -> bf16 A-fragment layout ----------
// frag index = t*16 + mt*4 + ks  (4 m-tiles, 4 k-steps of 32 over W=128)
__global__ void pack_x_kernel(const float* __restrict__ x, unsigned short* __restrict__ out) {
  int id = blockIdx.x * blockDim.x + threadIdx.x;   // total = T*16*512 = 8388608
  int frag = id >> 9;
  int t = frag >> 4;
  int mt = (frag >> 2) & 3;
  int ks = frag & 3;
  int lane = (id >> 4) & 31;
  int e = id & 15;
  int r = e >> 1, sub = e & 1;
  int k = ks * 32 + a_kpair(lane, r) + sub;
  int b = mt * 16 + (lane & 15);
  out[id] = f2bfu(x[(size_t)b * T_ * W_ + (size_t)t * W_ + k]);
}

__global__ void zero_f_kernel(float* __restrict__ p, int n) {
  int i = blockIdx.x * blockDim.x + threadIdx.x;
  if (i < n) p[i] = 0.0f;
}
__global__ void zero_us_kernel(unsigned short* __restrict__ p, int n) {
  int i = blockIdx.x * blockDim.x + threadIdx.x;
  if (i < n) p[i] = 0;
}

// ---------- fused LSTM step: z = [aleft | h_prev] @ Wk + b ; gate math ----------
// Grid: 32 blocks x 128 thr = 128 waves; wave -> (mt in 0..3, nt in 0..31).
// KX/KS are compile-time: the K loop fully unrolls, every fragment load is
// base + constant immediate offset, and the A-source select is static.
template <int KX, int KS>
__global__ __launch_bounds__(128) void lstm_step_kernel(
    const unsigned short* __restrict__ aleft, long aleft_frag_base,
    const unsigned short* __restrict__ h_prev,        // A-layout, frag = mt*16+ks
    const unsigned short* __restrict__ Bpack,
    const float* __restrict__ bias,
    float* __restrict__ c_state,
    unsigned short* __restrict__ h_out,               // A-layout, frag = mt*16+ks
    unsigned short* __restrict__ h2head, int t) {     // head A-layout (or null)
  int wave = (blockIdx.x << 2) + (threadIdx.x >> 5);
  int lane = threadIdx.x & 31;
  int mt = wave & 3;
  int nt = wave >> 2;
  int lofs = lane << 4;

  // per-wave base pointers; all in-loop offsets are immediates
  const unsigned short* aL = aleft + (((size_t)(aleft_frag_base + (long)mt * KX)) << 9) + lofs;
  const unsigned short* aH = h_prev + (((size_t)mt * 16) << 9) + lofs;
  const unsigned short* bB = Bpack + (((size_t)nt * KS) << 9) + lofs;

  v8f ai = {}, aj = {}, af = {}, ao = {};
#pragma unroll
  for (int ks = 0; ks < KS; ++ks) {
    v16bf a = (ks < KX) ? load_frag_off(aL, ks) : load_frag_off(aH, ks - KX);
    v16bf b0 = load_frag_off(bB, 0 * 32 * KS + ks);
    v16bf b1 = load_frag_off(bB, 1 * 32 * KS + ks);
    v16bf b2 = load_frag_off(bB, 2 * 32 * KS + ks);
    v16bf b3 = load_frag_off(bB, 3 * 32 * KS + ks);
    ai = __builtin_amdgcn_wmma_f32_16x16x32_bf16(false, a, false, b0, (short)0, ai, false, false);
    aj = __builtin_amdgcn_wmma_f32_16x16x32_bf16(false, a, false, b1, (short)0, aj, false, false);
    af = __builtin_amdgcn_wmma_f32_16x16x32_bf16(false, a, false, b2, (short)0, af, false, false);
    ao = __builtin_amdgcn_wmma_f32_16x16x32_bf16(false, a, false, b3, (short)0, ao, false, false);
  }

  // C/D layout: lane -> n = lane%16 ; vgpr r -> m = r (+8 for lanes 16-31)
  int n = lane & 15;
  int col = nt * 16 + n;
  int mb = (lane >= 16) ? 8 : 0;
  float bi = bias[col], bj = bias[H_ + col], bf = bias[2 * H_ + col], bo = bias[3 * H_ + col];

  // Invert the A 16x32 map for this lane's column (uniform per lane):
  int ks_a = col >> 5;
  int kk = col & 31;
  int half = (kk >> 3) & 1;                 // kk in {8..15, 24..31} -> upper lanes
  int kkp = kk - 8 * half;                  // now in {0..7} or {16..23}
  int rt = (kkp < 8) ? (kkp >> 1) : (4 + ((kkp - 16) >> 1));
  int ea = 2 * rt + (kkp & 1);              // element slot within lane's 16
  unsigned short* hr = h_out + (((size_t)(mt * 16 + ks_a)) << 9) + ea;
  int thi = (t & 15) + 16 * half;           // head-fragment lane (row-in-tile = t%16)

#pragma unroll
  for (int r = 0; r < 8; ++r) {
    int m = mb + r;
    int row = mt * 16 + m;
    float zi = ai[r] + bi, zj = aj[r] + bj, zf = af[r] + bf, zo = ao[r] + bo;
    size_t idx = (size_t)row * H_ + col;
    float cn = sigf(zf + FORGET_BIAS) * c_state[idx] + sigf(zi) * tanhf(zj);
    float hn = sigf(zo) * tanhf(cn);
    c_state[idx] = cn;
    unsigned short hb = f2bfu(hn);
    hr[(size_t)(m + 16 * half) << 4] = hb;  // recurrent A-layout store
    if (h2head) {                            // head A-layout: mtile = row*64 + t/16
      size_t frag_h = ((size_t)row * (T_ / 16) + (t >> 4)) * 16 + ks_a;
      h2head[(frag_h << 9) + ((size_t)thi << 4) + ea] = hb;
    }
  }
}

// ---------- head: logits = h2(bf16) @ Wout + bout ; fused sigmoid for pred ----------
// 65536 waves: wave -> (mtile 0..4095, nt 0..15); K = 512 -> 16 WMMA steps.
__global__ __launch_bounds__(256) void head_kernel(
    const unsigned short* __restrict__ h2head,     // A-layout, frag = mtile*16+ks
    const unsigned short* __restrict__ WoutPack,
    const float* __restrict__ bout,
    float* __restrict__ logits, float* __restrict__ pred) {
  int wave = (blockIdx.x << 3) + (threadIdx.x >> 5);
  int lane = threadIdx.x & 31;
  int nt = wave & 15;
  int mtile = wave >> 4;
  int lofs = lane << 4;
  const unsigned short* ap = h2head + (((size_t)mtile * 16) << 9) + lofs;
  const unsigned short* bp = WoutPack + (((size_t)nt * 16) << 9) + lofs;
  v8f acc = {};
#pragma unroll
  for (int ks = 0; ks < 16; ++ks) {
    v16bf a = load_frag_off(ap, ks);
    v16bf b = load_frag_off(bp, ks);
    acc = __builtin_amdgcn_wmma_f32_16x16x32_bf16(false, a, false, b, (short)0, acc, false, false);
  }
  int n = lane & 15, mb = (lane >= 16) ? 8 : 0;
  int col = nt * 16 + n;
  float bb = bout[col];
#pragma unroll
  for (int r = 0; r < 8; ++r) {
    int row = mtile * 16 + mb + r;
    float v = acc[r] + bb;
    logits[(size_t)row * C_ + col] = v;
    if ((row & (T_ - 1)) == (T_ - 1))  // last timestep rows: b*T + (T-1)
      pred[(size_t)(row >> 10) * C_ + col] = 1.0f / (1.0f + __expf(-v));
  }
}

// ---------- cost: per-row logsumexp - logit[label], wave32 shuffle reduction ----------
__global__ __launch_bounds__(256) void cost_rows_kernel(const float* __restrict__ logits,
                                                        const int* __restrict__ y_idx,
                                                        float* __restrict__ rowcost) {
  int wave = (blockIdx.x << 3) + (threadIdx.x >> 5);
  int lane = threadIdx.x & 31;
  const float* p = logits + (size_t)wave * C_;
  float m = -3.4e38f;
  for (int i = lane; i < C_; i += 32) m = fmaxf(m, p[i]);
#pragma unroll
  for (int off = 16; off; off >>= 1) m = fmaxf(m, __shfl_xor(m, off));
  float s = 0.0f;
  for (int i = lane; i < C_; i += 32) s += __expf(p[i] - m);
#pragma unroll
  for (int off = 16; off; off >>= 1) s += __shfl_xor(s, off);
  if (lane == 0) rowcost[wave] = (m + __logf(s)) - p[y_idx[wave]];
}

__global__ __launch_bounds__(256) void cost_reduce_kernel(const float* __restrict__ rowcost,
                                                          float* __restrict__ out, int n) {
  __shared__ float sm[256];
  float s = 0.0f;
  for (int i = threadIdx.x; i < n; i += 256) s += rowcost[i];
  sm[threadIdx.x] = s;
  __syncthreads();
  for (int w = 128; w; w >>= 1) {
    if ((int)threadIdx.x < w) sm[threadIdx.x] += sm[threadIdx.x + w];
    __syncthreads();
  }
  if (threadIdx.x == 0) *out = sm[0] / (float)n;
}

// ---------- launch ----------
extern "C" void kernel_launch(void* const* d_in, const int* in_sizes, int n_in,
                              void* d_out, int out_size, void* d_ws, size_t ws_size,
                              hipStream_t stream) {
  const float* x    = (const float*)d_in[0];
  const float* W1   = (const float*)d_in[1];
  const float* b1   = (const float*)d_in[2];
  const float* W2   = (const float*)d_in[3];
  const float* b2   = (const float*)d_in[4];
  const float* Wout = (const float*)d_in[5];
  const float* bout = (const float*)d_in[6];
  const int*   yidx = (const int*)d_in[7];

  float* logits = (float*)d_out;                       // [B*T, C]
  float* pred   = logits + (size_t)B_ * T_ * C_;       // [B, C]
  float* cost   = pred + (size_t)B_ * C_;              // [1]

  size_t off = 0;
  auto alloc = [&](size_t bytes) -> void* {
    void* p = (char*)d_ws + off;
    off += (bytes + 255) & ~(size_t)255;
    return p;
  };
  unsigned short* w1p    = (unsigned short*)alloc((size_t)(W_ + H_) * 4 * H_ * 2); // 128nt*20ks
  unsigned short* w2p    = (unsigned short*)alloc((size_t)(2 * H_) * 4 * H_ * 2);  // 128nt*32ks
  unsigned short* wop    = (unsigned short*)alloc((size_t)H_ * C_ * 2);            // 16nt*16ks
  unsigned short* xpack  = (unsigned short*)alloc((size_t)B_ * T_ * W_ * 2);       // A-frag x
  unsigned short* h2head = (unsigned short*)alloc((size_t)B_ * T_ * H_ * 2);       // A-frag h2
  unsigned short* hus    = (unsigned short*)alloc((size_t)4 * B_ * H_ * 2);  // h1a,h2a,h1b,h2b
  float*          cst    = (float*)alloc((size_t)2 * B_ * H_ * 4);           // c1,c2
  float*          rowcost = (float*)alloc((size_t)B_ * T_ * 4);

  unsigned short* h1a = hus;
  unsigned short* h2a = hus + 1 * B_ * H_;
  unsigned short* h1b = hus + 2 * B_ * H_;
  unsigned short* h2b = hus + 3 * B_ * H_;
  float* c1 = cst;
  float* c2 = cst + B_ * H_;

  // zero initial states (h1a, h2a contiguous; c1, c2 contiguous)
  zero_us_kernel<<<(2 * B_ * H_ + 255) / 256, 256, 0, stream>>>(hus, 2 * B_ * H_);
  zero_f_kernel<<<(2 * B_ * H_ + 255) / 256, 256, 0, stream>>>(cst, 2 * B_ * H_);

  // pre-swizzle weights + x into WMMA fragment layouts
  {
    int t1 = 128 * 20 * 512, t2 = 128 * 32 * 512, t3 = 16 * 16 * 512;
    pack_b_kernel<<<(t1 + 255) / 256, 256, 0, stream>>>(W1, w1p, W_ + H_, 4 * H_, 20);
    pack_b_kernel<<<(t2 + 255) / 256, 256, 0, stream>>>(W2, w2p, 2 * H_, 4 * H_, 32);
    pack_b_kernel<<<(t3 + 255) / 256, 256, 0, stream>>>(Wout, wop, H_, C_, 16);
    int tx = B_ * T_ * W_;
    pack_x_kernel<<<(tx + 255) / 256, 256, 0, stream>>>(x, xpack);
  }

  // sequential recurrence: 1024 steps x 2 layers, ping-pong h buffers
  for (int t = 0; t < T_; ++t) {
    unsigned short* h1_prev = (t & 1) ? h1b : h1a;
    unsigned short* h1_cur  = (t & 1) ? h1a : h1b;
    unsigned short* h2_prev = (t & 1) ? h2b : h2a;
    unsigned short* h2_cur  = (t & 1) ? h2a : h2b;
    lstm_step_kernel<4, 20><<<32, 128, 0, stream>>>(
        xpack, (long)t * 16,
        h1_prev, w1p, b1, c1, h1_cur,
        (unsigned short*)nullptr, t);
    lstm_step_kernel<16, 32><<<32, 128, 0, stream>>>(
        h1_cur, 0L,
        h2_prev, w2p, b2, c2, h2_cur,
        h2head, t);
  }

  // dense head + prediction
  head_kernel<<<8192, 256, 0, stream>>>(h2head, wop, bout, logits, pred);

  // softmax cross-entropy cost (deterministic two-stage reduction)
  cost_rows_kernel<<<8192, 256, 0, stream>>>(logits, yidx, rowcost);
  cost_reduce_kernel<<<1, 256, 0, stream>>>(rowcost, cost, B_ * T_);
}
